// GroupedQueryAttention_22978075034352
// MI455X (gfx1250) — matrix operand order repaired and checked by reading, not compiled, not run on hardware
//
#include <hip/hip_runtime.h>

#define DM   1024
#define NHQ  16
#define NKV  4
#define DK   64
#define SQ   2048
#define BB   2

typedef __bf16 bf16;
typedef __attribute__((ext_vector_type(16))) __bf16 v16bf;
typedef __attribute__((ext_vector_type(8)))  __bf16 bf16x8;
typedef __attribute__((ext_vector_type(8)))  float  v8f;
typedef __attribute__((ext_vector_type(4)))  float  f32x4;

union V16U { bf16x8 h[2]; v16bf v; };

// ---- WMMA fragment loaders (wave32 layouts, ISA 7.12.2) ----------------
// A-matrix 16x32 bf16: lane m = lane&15; K runs [8*(lane>>4) .. +7] and +16.
__device__ inline v16bf load_a_frag(const bf16* base, int rowStride, int lane) {
    const bf16* p = base + (size_t)(lane & 15) * rowStride + ((lane >> 4) * 8);
    V16U u;
    u.h[0] = *(const bf16x8*)(p);
    u.h[1] = *(const bf16x8*)(p + 16);
    return u.v;
}
// B-matrix 32x16 bf16: lane n = lane&15; K runs [8*(lane>>4) .. +7] and +16.
__device__ inline v16bf load_b_frag(const bf16* base, int rowStride, int lane) {
    const bf16* p = base + (size_t)(lane & 15) * rowStride + ((lane >> 4) * 8);
    V16U u;
    u.h[0] = *(const bf16x8*)(p);
    u.h[1] = *(const bf16x8*)(p + 16);
    return u.v;
}

#define WMMA_BF16(A, B, C) \
    __builtin_amdgcn_wmma_f32_16x16x32_bf16(false, (A), false, (B), (short)0, (C), false, false)

// ---- Kernel 1: projection GEMM  out[b,h,s,d] = sum_k X[m,k]*W[n,k] -----
__global__ __launch_bounds__(256)
void proj_kernel(const float* __restrict__ X, const float* __restrict__ W,
                 bf16* __restrict__ out, int nheads)
{
    __shared__ __align__(16) bf16 As[64 * 32];
    __shared__ __align__(16) bf16 Bs[64 * 32];

    const int m0 = blockIdx.x * 64;
    const int n0 = blockIdx.y * 64;
    const int t    = threadIdx.x;
    const int lane = t & 31;
    const int w    = t >> 5;          // 8 waves
    const int wm   = (w & 3) * 16;    // 4 row-stripes of 16
    const int wn   = (w >> 2) * 32;   // 2 col-stripes of 32

    const int arow = t >> 2;          // 0..63
    const int acol = (t & 3) * 8;     // 0,8,16,24

    v8f acc0 = {0,0,0,0,0,0,0,0};
    v8f acc1 = {0,0,0,0,0,0,0,0};

    for (int k0 = 0; k0 < DM; k0 += 32) {
        __syncthreads();
        {   // stage A tile (f32 -> bf16)
            const float* src = X + (size_t)(m0 + arow) * DM + k0 + acol;
            f32x4 f0 = *(const f32x4*)(src);
            f32x4 f1 = *(const f32x4*)(src + 4);
            bf16x8 h;
            h[0]=(bf16)f0[0]; h[1]=(bf16)f0[1]; h[2]=(bf16)f0[2]; h[3]=(bf16)f0[3];
            h[4]=(bf16)f1[0]; h[5]=(bf16)f1[1]; h[6]=(bf16)f1[2]; h[7]=(bf16)f1[3];
            *(bf16x8*)(As + arow * 32 + acol) = h;
        }
        {   // stage B tile = W rows (f32 -> bf16)
            const float* src = W + (size_t)(n0 + arow) * DM + k0 + acol;
            f32x4 f0 = *(const f32x4*)(src);
            f32x4 f1 = *(const f32x4*)(src + 4);
            bf16x8 h;
            h[0]=(bf16)f0[0]; h[1]=(bf16)f0[1]; h[2]=(bf16)f0[2]; h[3]=(bf16)f0[3];
            h[4]=(bf16)f1[0]; h[5]=(bf16)f1[1]; h[6]=(bf16)f1[2]; h[7]=(bf16)f1[3];
            *(bf16x8*)(Bs + arow * 32 + acol) = h;
        }
        if (k0 + 32 < DM)
            __builtin_prefetch(X + (size_t)(m0 + arow) * DM + k0 + 32 + acol, 0, 1);
        __syncthreads();

        v16bf a  = load_a_frag(As + wm * 32, 32, lane);
        v16bf b0 = load_b_frag(Bs + wn * 32, 32, lane);
        v16bf b1 = load_b_frag(Bs + (wn + 16) * 32, 32, lane);
        acc0 = WMMA_BF16(a, b0, acc0);
        acc1 = WMMA_BF16(a, b1, acc1);
    }

    // C layout: m = r + 8*(lane>>4), n = lane&15. Store to (b,h,s,d).
    const int nl = lane & 15;
    const int mh = (lane >> 4) * 8;
    #pragma unroll
    for (int f = 0; f < 2; ++f) {
        const int gn = n0 + wn + f * 16 + nl;
        const int h  = gn >> 6;
        const int d  = gn & 63;
        #pragma unroll
        for (int r = 0; r < 8; ++r) {
            const int gm = m0 + wm + mh + r;
            const int b  = gm >> 11;
            const int s  = gm & (SQ - 1);
            const float val = f ? acc1[r] : acc0[r];
            out[(((size_t)b * nheads + h) * SQ + s) * DK + d] = (bf16)val;
        }
    }
}

// ---- Kernel 2: RoPE in-place on (b, h, s, d) bf16 ----------------------
__global__ void rope_kernel(bf16* __restrict__ q)
{
    const int t = blockIdx.x * blockDim.x + threadIdx.x;
    const int i  = t & 31;                 // pair index 0..31
    const int s  = (t >> 5) & (SQ - 1);
    const int bh = t >> 16;                // 32*2048 = 65536
    bf16* row = q + ((size_t)bh * SQ + s) * DK;
    // inv_freq = 10000^(-2i/64) ; ln(10000) = 9.210340372
    const float f = __expf(-(float)(2 * i) * (9.210340371976184f / 64.0f));
    const float ang = (float)s * f;
    float sn, cs;
    __sincosf(ang, &sn, &cs);
    const float x1 = (float)row[i];
    const float x2 = (float)row[i + 32];
    row[i]      = (bf16)(x1 * cs - x2 * sn);
    row[i + 32] = (bf16)(x2 * cs + x1 * sn);
}

// ---- Kernel 3: causal flash attention (GQA) ----------------------------
__global__ __launch_bounds__(128)
void flash_kernel(const bf16* __restrict__ Qb, const bf16* __restrict__ Kb,
                  const bf16* __restrict__ Vb, bf16* __restrict__ AO)
{
    __shared__ __align__(16) bf16 Ks [64 * 64];  // [k][d]
    __shared__ __align__(16) bf16 VsT[64 * 64];  // [d][k]
    __shared__ __align__(16) bf16 Ps [64 * 64];  // [q][k]

    const int qt  = blockIdx.x;
    const int hh  = blockIdx.y;
    const int b   = blockIdx.z;
    const int kvh = hh >> 2;               // GROUP = 4
    const int t    = threadIdx.x;
    const int lane = t & 31;
    const int w    = t >> 5;               // 4 waves: 16 q-rows each
    const int q0   = qt * 64;

    const bf16* qbase = Qb + (((size_t)b * NHQ + hh) * SQ + q0 + w * 16) * DK;
    const v16bf aQ0 = load_a_frag(qbase,      DK, lane);
    const v16bf aQ1 = load_a_frag(qbase + 32, DK, lane);

    const bf16* kbase = Kb + (((size_t)b * NKV + kvh) * SQ) * DK;
    const bf16* vbase = Vb + (((size_t)b * NKV + kvh) * SQ) * DK;

    float mrow[8], lrow[8];
    v8f o[4];
    #pragma unroll
    for (int dt = 0; dt < 4; ++dt)
        #pragma unroll
        for (int r = 0; r < 8; ++r) o[dt][r] = 0.0f;
    #pragma unroll
    for (int r = 0; r < 8; ++r) { mrow[r] = -3.0e38f; lrow[r] = 0.0f; }

    const int nl = lane & 15;
    const int mh = (lane >> 4) * 8;
    const int qrow_base = q0 + w * 16 + mh;

    for (int jt = 0; jt <= qt; ++jt) {
        const int j0 = jt * 64;
        __syncthreads();
        {   // stage K[k][d] and V^T[d][k]; thread: row = t>>1, 32 cols
            const int krow = t >> 1;
            const int dc   = (t & 1) * 32;
            const bf16* ks = kbase + (size_t)(j0 + krow) * DK + dc;
            const bf16* vs = vbase + (size_t)(j0 + krow) * DK + dc;
            if (jt < qt) __builtin_prefetch(ks + 64 * DK, 0, 1);
            #pragma unroll
            for (int u = 0; u < 32; u += 8) {
                bf16x8 kv = *(const bf16x8*)(ks + u);
                *(bf16x8*)(Ks + krow * 64 + dc + u) = kv;
                bf16x8 vv = *(const bf16x8*)(vs + u);
                #pragma unroll
                for (int e = 0; e < 8; ++e)
                    VsT[(dc + u + e) * 64 + krow] = vv[e];
            }
        }
        __syncthreads();

        // S = Q K^T  (4 n-tiles x 2 k-steps)
        v8f sc[4];
        #pragma unroll
        for (int nt = 0; nt < 4; ++nt) {
            v8f a = {0,0,0,0,0,0,0,0};
            v16bf bk0 = load_b_frag(Ks + nt * 16 * 64,      64, lane);
            v16bf bk1 = load_b_frag(Ks + nt * 16 * 64 + 32, 64, lane);
            a = WMMA_BF16(aQ0, bk0, a);
            a = WMMA_BF16(aQ1, bk1, a);
            sc[nt] = a;
        }
        // scale + causal mask (matches reference: masked = -10000.0)
        #pragma unroll
        for (int nt = 0; nt < 4; ++nt) {
            const int kcol = j0 + nt * 16 + nl;
            #pragma unroll
            for (int r = 0; r < 8; ++r) {
                const float v = sc[nt][r] * 0.125f;
                sc[nt][r] = (kcol > qrow_base + r) ? -10000.0f : v;
            }
        }
        // online softmax per row (rows live in 16-lane groups)
        #pragma unroll
        for (int r = 0; r < 8; ++r) {
            float rm = fmaxf(fmaxf(sc[0][r], sc[1][r]), fmaxf(sc[2][r], sc[3][r]));
            rm = fmaxf(rm, __shfl_xor(rm, 1));
            rm = fmaxf(rm, __shfl_xor(rm, 2));
            rm = fmaxf(rm, __shfl_xor(rm, 4));
            rm = fmaxf(rm, __shfl_xor(rm, 8));
            const float mnew  = fmaxf(mrow[r], rm);
            const float alpha = __expf(mrow[r] - mnew);
            float rs = 0.0f;
            #pragma unroll
            for (int nt = 0; nt < 4; ++nt) {
                const float p = __expf(sc[nt][r] - mnew);
                sc[nt][r] = p;
                rs += p;
            }
            rs += __shfl_xor(rs, 1);
            rs += __shfl_xor(rs, 2);
            rs += __shfl_xor(rs, 4);
            rs += __shfl_xor(rs, 8);
            lrow[r] = lrow[r] * alpha + rs;
            mrow[r] = mnew;
            #pragma unroll
            for (int dt = 0; dt < 4; ++dt) o[dt][r] *= alpha;
        }
        // P -> LDS (C-layout -> A-layout via shared memory)
        #pragma unroll
        for (int nt = 0; nt < 4; ++nt)
            #pragma unroll
            for (int r = 0; r < 8; ++r)
                Ps[(w * 16 + mh + r) * 64 + nt * 16 + nl] = (bf16)sc[nt][r];
        __syncthreads();

        const v16bf aP0 = load_a_frag(Ps + (w * 16) * 64,      64, lane);
        const v16bf aP1 = load_a_frag(Ps + (w * 16) * 64 + 32, 64, lane);
        #pragma unroll
        for (int dt = 0; dt < 4; ++dt) {
            v16bf bv0 = load_b_frag(VsT + dt * 16 * 64,      64, lane);
            v16bf bv1 = load_b_frag(VsT + dt * 16 * 64 + 32, 64, lane);
            o[dt] = WMMA_BF16(aP0, bv0, o[dt]);
            o[dt] = WMMA_BF16(aP1, bv1, o[dt]);
        }
    }

    // finalize: O /= l ; store to (b, s, hh, d)  == (B,S,1024) row-major
    #pragma unroll
    for (int dt = 0; dt < 4; ++dt) {
        const int d = dt * 16 + nl;
        #pragma unroll
        for (int r = 0; r < 8; ++r) {
            const int q = qrow_base + r;
            const float val = o[dt][r] / lrow[r];
            AO[(((size_t)b * SQ + q) * NHQ + hh) * DK + d] = (bf16)val;
        }
    }
}

// ---- Kernel 4: output projection  out = AO @ Wo^T  (f32 result) --------
__global__ __launch_bounds__(256)
void oproj_kernel(const bf16* __restrict__ A, const float* __restrict__ W,
                  float* __restrict__ out)
{
    __shared__ __align__(16) bf16 As[64 * 32];
    __shared__ __align__(16) bf16 Bs[64 * 32];

    const int m0 = blockIdx.x * 64;
    const int n0 = blockIdx.y * 64;
    const int t    = threadIdx.x;
    const int lane = t & 31;
    const int w    = t >> 5;
    const int wm   = (w & 3) * 16;
    const int wn   = (w >> 2) * 32;
    const int arow = t >> 2;
    const int acol = (t & 3) * 8;

    v8f acc0 = {0,0,0,0,0,0,0,0};
    v8f acc1 = {0,0,0,0,0,0,0,0};

    for (int k0 = 0; k0 < DM; k0 += 32) {
        __syncthreads();
        {   // stage A (already bf16)
            bf16x8 av = *(const bf16x8*)(A + (size_t)(m0 + arow) * DM + k0 + acol);
            *(bf16x8*)(As + arow * 32 + acol) = av;
        }
        {   // stage Wo (f32 -> bf16)
            const float* src = W + (size_t)(n0 + arow) * DM + k0 + acol;
            f32x4 f0 = *(const f32x4*)(src);
            f32x4 f1 = *(const f32x4*)(src + 4);
            bf16x8 h;
            h[0]=(bf16)f0[0]; h[1]=(bf16)f0[1]; h[2]=(bf16)f0[2]; h[3]=(bf16)f0[3];
            h[4]=(bf16)f1[0]; h[5]=(bf16)f1[1]; h[6]=(bf16)f1[2]; h[7]=(bf16)f1[3];
            *(bf16x8*)(Bs + arow * 32 + acol) = h;
        }
        __syncthreads();

        v16bf a  = load_a_frag(As + wm * 32, 32, lane);
        v16bf b0 = load_b_frag(Bs + wn * 32, 32, lane);
        v16bf b1 = load_b_frag(Bs + (wn + 16) * 32, 32, lane);
        acc0 = WMMA_BF16(a, b0, acc0);
        acc1 = WMMA_BF16(a, b1, acc1);
    }

    const int nl = lane & 15;
    const int mh = (lane >> 4) * 8;
    #pragma unroll
    for (int f = 0; f < 2; ++f) {
        const int gn = n0 + wn + f * 16 + nl;
        #pragma unroll
        for (int r = 0; r < 8; ++r) {
            const int gm = m0 + wm + mh + r;
            out[(size_t)gm * DM + gn] = f ? acc1[r] : acc0[r];
        }
    }
}

// ------------------------------------------------------------------------
extern "C" void kernel_launch(void* const* d_in, const int* in_sizes, int n_in,
                              void* d_out, int out_size, void* d_ws, size_t ws_size,
                              hipStream_t stream)
{
    (void)in_sizes; (void)n_in; (void)out_size; (void)ws_size;
    const float* q  = (const float*)d_in[0];
    const float* k  = (const float*)d_in[1];
    const float* v  = (const float*)d_in[2];
    const float* Wq = (const float*)d_in[3];
    const float* Wk = (const float*)d_in[4];
    const float* Wv = (const float*)d_in[5];
    const float* Wo = (const float*)d_in[6];
    float* out = (float*)d_out;

    char* ws = (char*)d_ws;
    bf16* Qb = (bf16*)(ws);                       //  8 MB: (B,16,S,64)
    bf16* Kb = (bf16*)(ws + ((size_t)8  << 20));  //  2 MB: (B, 4,S,64)
    bf16* Vb = (bf16*)(ws + ((size_t)10 << 20));  //  2 MB: (B, 4,S,64)
    bf16* AO = (bf16*)(ws + ((size_t)12 << 20));  //  8 MB: (B,S,1024)

    // QKV projections (M = B*S = 4096 rows)
    proj_kernel<<<dim3(64, 16), dim3(256), 0, stream>>>(q, Wq, Qb, NHQ);
    proj_kernel<<<dim3(64, 4),  dim3(256), 0, stream>>>(k, Wk, Kb, NKV);
    proj_kernel<<<dim3(64, 4),  dim3(256), 0, stream>>>(v, Wv, Vb, NKV);

    // RoPE on Q and K
    rope_kernel<<<(BB * NHQ * SQ * 32) / 256, 256, 0, stream>>>(Qb);
    rope_kernel<<<(BB * NKV * SQ * 32) / 256, 256, 0, stream>>>(Kb);

    // Flash attention: 32 q-tiles x 16 heads x 2 batches
    flash_kernel<<<dim3(SQ / 64, NHQ, BB), dim3(128), 0, stream>>>(Qb, Kb, Vb, AO);

    // Output projection -> f32
    oproj_kernel<<<dim3(64, 16), dim3(256), 0, stream>>>(AO, Wo, out);
}
